// ModulatedConv2d_13314398618224
// MI455X (gfx1250) — matrix operand order repaired and checked by reading, not compiled, not run on hardware
//
#include <hip/hip_runtime.h>

typedef __attribute__((ext_vector_type(16))) __bf16 v16bf;
typedef __attribute__((ext_vector_type(8)))  float  v8f;

#define CIN   512
#define COUT  512
#define SDIM  512
#define HW    4096       // 64*64
#define RDIM  4608       // CIN*9, implicit-GEMM K; ordered as tap*512 + ci
#define LDST  40         // LDS row stride in bf16 (80B): conflict-free b128 reads
#define KITER 144        // RDIM / 32

// ---------------------------------------------------------------------------
// 1) style projection: s[b][i] = style[b] . proj_w[i] + proj_b[i]
// ---------------------------------------------------------------------------
__global__ __launch_bounds__(256) void style_proj_kernel(
    const float* __restrict__ style, const float* __restrict__ proj_w,
    const float* __restrict__ proj_b, float* __restrict__ s)
{
    int id = blockIdx.x * 256 + threadIdx.x;   // B*CIN threads
    int b = id >> 9, i = id & 511;
    float acc = proj_b[i];
    const float* st = style + (size_t)b * SDIM;
    const float* pw = proj_w + (size_t)i * SDIM;
    for (int d = 0; d < SDIM; ++d) acc += st[d] * pw[d];
    s[id] = acc;
}

// ---------------------------------------------------------------------------
// 2) weight fp32 -> bf16, transposed to [o][tap][ci] (tap-outer K ordering)
// ---------------------------------------------------------------------------
__global__ __launch_bounds__(256) void wconv_kernel(
    const float* __restrict__ w, __bf16* __restrict__ wbf)
{
    int id = blockIdx.x * 256 + threadIdx.x;   // COUT*RDIM
    int o   = id / RDIM;
    int rem = id - o * RDIM;
    int ci  = rem / 9;
    int tap = rem - ci * 9;
    wbf[(size_t)o * RDIM + tap * CIN + ci] = (__bf16)w[id];
}

// ---------------------------------------------------------------------------
// 3) demod[b][o] = rsqrt( sum_{i,k} (w[o,i,k]*s[b,i])^2 + 1e-8 )
// ---------------------------------------------------------------------------
__global__ __launch_bounds__(256) void demod_kernel(
    const float* __restrict__ s, const float* __restrict__ w,
    float* __restrict__ demod)
{
    int id = blockIdx.x * 256 + threadIdx.x;   // B*COUT threads
    int b = id >> 9, o = id & 511;
    const float* sb = s + (size_t)b * CIN;
    const float* wo = w + (size_t)o * RDIM;
    float acc = 1e-8f;
    for (int i = 0; i < CIN; ++i) {
        float si = sb[i];
        float si2 = si * si;
        #pragma unroll
        for (int k = 0; k < 9; ++k) { float wv = wo[i * 9 + k]; acc += wv * wv * si2; }
    }
    demod[id] = rsqrtf(acc);
}

// ---------------------------------------------------------------------------
// 4) pre-modulated NHWC bf16 activations: xm[b][p][ci] = x[b][ci][p]*s[b][ci]
// ---------------------------------------------------------------------------
__global__ __launch_bounds__(256) void xmod_kernel(
    const float* __restrict__ x, const float* __restrict__ s,
    __bf16* __restrict__ xm)
{
    __shared__ float tile[32][33];
    __shared__ float sv[32];
    int tx = threadIdx.x & 31;
    int ty = threadIdx.x >> 5;      // 0..7
    int b   = blockIdx.z;
    int ciB = blockIdx.y * 32;
    int pB  = blockIdx.x * 32;
    if (threadIdx.x < 32) sv[threadIdx.x] = s[b * CIN + ciB + threadIdx.x];
    __syncthreads();
    #pragma unroll
    for (int k = 0; k < 4; ++k) {
        int ci = ty + k * 8;
        tile[ci][tx] = x[((size_t)(b * CIN + ciB + ci)) * HW + pB + tx] * sv[ci];
    }
    __syncthreads();
    #pragma unroll
    for (int k = 0; k < 4; ++k) {
        int p = ty + k * 8;
        xm[((size_t)(b * HW + pB + p)) * CIN + ciB + tx] = (__bf16)tile[tx][p];
    }
}

// ---------------------------------------------------------------------------
// tiny zeroed region used as async source for halo lanes
// ---------------------------------------------------------------------------
__global__ void zero_kernel(float* __restrict__ z)
{
    z[threadIdx.x] = 0.0f;   // 64 threads -> 256B of zeros
}

// ---------------------------------------------------------------------------
// 5) main implicit-GEMM modulated conv, bf16 WMMA, f32 accumulate
//    block: 256 threads (8 wave32), tile 128(O) x 128(HW)
//    K-loop: 144 chunks of 32, triple-buffered async direct-to-LDS staging
// ---------------------------------------------------------------------------
union FragU { uint4 q[2]; v16bf v; };

__device__ __forceinline__ void async_copy32B(unsigned ldsAddr,
                                              unsigned long long gAddr)
{
    // global -> LDS copy, 16B per lane; offset applies to both.
    asm volatile("global_load_async_to_lds_b128 %0, %1, off\n\t"
                 "global_load_async_to_lds_b128 %0, %1, off offset:16"
                 :: "v"(ldsAddr), "v"(gAddr) : "memory");
}

template<int N>
__device__ __forceinline__ void wait_async()
{
    asm volatile("s_wait_asynccnt %0" :: "i"(N) : "memory");
}

template<bool PREMOD>
__global__ __launch_bounds__(256) void modconv_wmma_kernel(
    const float*  __restrict__ x,     const __bf16* __restrict__ xm,
    const float*  __restrict__ noise, const float*  __restrict__ nstr,
    const float*  __restrict__ bias,  const float*  __restrict__ s,
    const float*  __restrict__ demod, const __bf16* __restrict__ wbf,
    const __bf16* __restrict__ zbuf,  float*        __restrict__ out)
{
    __shared__ __align__(16) __bf16 sA[3][128 * LDST];
    __shared__ __align__(16) __bf16 sB[3][128 * LDST];
    __shared__ float sS[CIN];

    constexpr int APS = PREMOD ? 4 : 2;   // async instructions per stage

    const int tid    = threadIdx.x;
    const int lane   = tid & 31;
    const int wave   = tid >> 5;     // 0..7
    const int waveO  = wave & 3;     // 32-row O sub-tile
    const int waveP  = wave >> 2;    // 64-col HW sub-tile
    const int laneLo = lane & 15;
    const int hiHalf = lane >> 4;

    const int b      = blockIdx.z;
    const int oBlock = blockIdx.y * 128;
    const int pBlock = blockIdx.x * 128;

    // cooperative-load roles
    const int aRow  = tid >> 1;      // 0..127 (O row)
    const int aHalf = tid & 1;       // 16 bf16 each
    const int bP    = tid >> 1;      // 0..127 (HW position)
    const int bHalf = tid & 1;
    const int h0 = (pBlock + bP) >> 6;
    const int w0 = (pBlock + bP) & 63;

    if constexpr (!PREMOD) {
        sS[tid]       = s[b * CIN + tid];
        sS[tid + 256] = s[b * CIN + tid + 256];
    }

    const __bf16* wrow = wbf + (size_t)(oBlock + aRow) * RDIM + aHalf * 16;
    const __bf16* xmb  = xm  + (size_t)b * HW * CIN;
    const float*  xb   = x   + (size_t)b * CIN * HW;

    v8f acc[8] = {};   // [m(2)][t(4)] 16x16 f32 tiles

    // ---- stage K-chunk `it` into LDS buffer `buf` (branchless, APS asyncs) --
    auto stage = [&](int it, int buf) {
        // A tile: K ordering is linear -> chunk address is wrow + it*32
        {
            unsigned ldsA = (unsigned)(size_t)(sA[buf] + aRow * LDST + aHalf * 16);
            const __bf16* ga = wrow + it * 32;
            __builtin_prefetch(ga + 32, 0, 3);   // near-cache prefetch of next chunk
            async_copy32B(ldsA, (unsigned long long)(size_t)ga);
        }
        if constexpr (PREMOD) {
            int tap = it >> 4;                    // uniform
            int cc  = (it & 15) << 5;             // uniform
            int kh  = tap / 3;
            int kw  = tap - kh * 3;
            int h = h0 + kh - 1, w = w0 + kw - 1;
            bool valid = ((unsigned)h < 64u) & ((unsigned)w < 64u);
            const __bf16* gb = xmb + (size_t)(h * 64 + w) * CIN + cc + bHalf * 16;
            // halo lanes read from the zeroed scratch region: count stays APS
            unsigned long long ga = valid ? (unsigned long long)(size_t)gb
                                          : (unsigned long long)(size_t)zbuf;
            unsigned ldsB = (unsigned)(size_t)(sB[buf] + bP * LDST + bHalf * 16);
            async_copy32B(ldsB, ga);
        } else {
            int tap = it >> 4;
            int cc  = (it & 15) << 5;
            int kh  = tap / 3;
            int kw  = tap - kh * 3;
            int h = h0 + kh - 1, w = w0 + kw - 1;
            bool valid = ((unsigned)h < 64u) & ((unsigned)w < 64u);
            union { __bf16 e[16]; uint4 q[2]; } tmp;
            int srcP = h * 64 + w;
            #pragma unroll
            for (int j = 0; j < 16; ++j) {
                int ci = cc + bHalf * 16 + j;
                float v = 0.0f;
                if (valid) v = xb[(size_t)ci * HW + srcP] * sS[ci];
                tmp.e[j] = (__bf16)v;
            }
            uint4* bd = (uint4*)(sB[buf] + bP * LDST + bHalf * 16);
            bd[0] = tmp.q[0];
            bd[1] = tmp.q[1];
        }
    };

    auto compute = [&](int buf) {
        // A layout: lanes 0-15 row M, K{0..7,16..23}; lanes 16-31 K{8..15,24..31}
        FragU a0f, a1f;
        const __bf16* ab = sA[buf] + (waveO * 32 + laneLo) * LDST + hiHalf * 8;
        a0f.q[0] = *(const uint4*)(ab);
        a0f.q[1] = *(const uint4*)(ab + 16);
        a1f.q[0] = *(const uint4*)(ab + 16 * LDST);
        a1f.q[1] = *(const uint4*)(ab + 16 * LDST + 16);
        #pragma unroll
        for (int t = 0; t < 4; ++t) {
            // B layout: lanes 0-15 col N K{0..7,16..23}; lanes 16-31 col N K{8..15,24..31}
            FragU bf;
            const __bf16* bb =
                sB[buf] + (waveP * 64 + t * 16 + laneLo) * LDST + hiHalf * 8;
            bf.q[0] = *(const uint4*)(bb);
            bf.q[1] = *(const uint4*)(bb + 16);
            acc[t] = __builtin_amdgcn_wmma_f32_16x16x32_bf16(
                false, a0f.v, false, bf.v, (short)0, acc[t], false, false);
            acc[4 + t] = __builtin_amdgcn_wmma_f32_16x16x32_bf16(
                false, a1f.v, false, bf.v, (short)0, acc[4 + t], false, false);
        }
    };

    // one pipeline step: stage it+2 (2 iterations ahead), compute it,
    // then wait for everything except the stage just issued.
    auto step = [&](int it, int cbuf, int sbuf2) {
        const bool staged = (it + 2) < KITER;
        if (staged) stage(it + 2, sbuf2);
        compute(cbuf);
        if (staged) wait_async<APS>();
        else        wait_async<0>();
        __syncthreads();
    };

    // prologue: fill buffers 0 and 1; ensure buffer 0 is complete
    stage(0, 0);
    stage(1, 1);
    wait_async<APS>();
    __syncthreads();

    for (int base = 0; base < KITER; base += 3) {
        step(base + 0, 0, 2);
        step(base + 1, 1, 0);
        step(base + 2, 2, 1);
    }

    // epilogue: demod, noise, bias; C layout M = v + 8*hiHalf, N = laneLo
    const float nsv = nstr[0];
    #pragma unroll
    for (int t = 0; t < 4; ++t) {
        int p = pBlock + waveP * 64 + t * 16 + laneLo;
        float nz = noise[(size_t)b * HW + p] * nsv;
        #pragma unroll
        for (int m = 0; m < 2; ++m) {
            #pragma unroll
            for (int v = 0; v < 8; ++v) {
                int o = oBlock + waveO * 32 + m * 16 + v + hiHalf * 8;
                float dm = demod[b * COUT + o];
                out[((size_t)(b * COUT + o)) * HW + p] =
                    acc[m * 4 + t][v] * dm + nz + bias[o];
            }
        }
    }
}

// ---------------------------------------------------------------------------
extern "C" void kernel_launch(void* const* d_in, const int* in_sizes, int n_in,
                              void* d_out, int out_size, void* d_ws, size_t ws_size,
                              hipStream_t stream)
{
    const float* x      = (const float*)d_in[0];
    const float* style  = (const float*)d_in[1];
    const float* noise  = (const float*)d_in[2];
    const float* weight = (const float*)d_in[3];
    const float* proj_w = (const float*)d_in[4];
    const float* proj_b = (const float*)d_in[5];
    const float* nstr   = (const float*)d_in[6];
    const float* bias   = (const float*)d_in[7];
    float* out = (float*)d_out;

    const int B = in_sizes[1] / SDIM;   // 16

    char*   ws    = (char*)d_ws;
    float*  sbuf  = (float*)ws;                      // B*CIN f32    (32 KB)
    float*  demod = (float*)(ws + 64 * 1024);        // B*COUT f32   (32 KB)
    __bf16* wbf   = (__bf16*)(ws + 128 * 1024);      // COUT*RDIM bf16 (~4.7 MB)
    const size_t ZERO_OFF = 6u * 1024 * 1024;
    __bf16* zbuf  = (__bf16*)(ws + ZERO_OFF);        // 256B zeroed halo source
    const size_t XM_OFF = 8u * 1024 * 1024;
    __bf16* xm    = (__bf16*)(ws + XM_OFF);          // B*HW*CIN bf16 (64 MB)

    const size_t need = XM_OFF + (size_t)B * HW * CIN * sizeof(__bf16);
    const bool premod = ws_size >= need;

    style_proj_kernel<<<(B * CIN) / 256, 256, 0, stream>>>(style, proj_w, proj_b, sbuf);
    wconv_kernel<<<(COUT * RDIM) / 256, 256, 0, stream>>>(weight, wbf);
    demod_kernel<<<(B * COUT) / 256, 256, 0, stream>>>(sbuf, weight, demod);

    dim3 grid(HW / 128, COUT / 128, B);
    if (premod) {
        zero_kernel<<<1, 64, 0, stream>>>((float*)zbuf);
        dim3 tgrid(HW / 32, CIN / 32, B);
        xmod_kernel<<<tgrid, 256, 0, stream>>>(x, sbuf, xm);
        modconv_wmma_kernel<true><<<grid, 256, 0, stream>>>(
            x, xm, noise, nstr, bias, sbuf, demod, wbf, zbuf, out);
    } else {
        modconv_wmma_kernel<false><<<grid, 256, 0, stream>>>(
            x, xm, noise, nstr, bias, sbuf, demod, wbf, zbuf, out);
    }
}
